// TensorNetwork_90958817395305
// MI455X (gfx1250) — compile-verified, matrix operand order checked by reading, not run on hardware
//
#include <hip/hip_runtime.h>

typedef __attribute__((ext_vector_type(16))) _Float16 v16h;
typedef __attribute__((ext_vector_type(8)))  float    v8f;

constexpr int kH   = 128;
constexpr int kRBF = 32;
constexpr int kN   = 10000;
constexpr int kE   = 80000;
constexpr int kN9H = kN * 9 * kH;

// GEMM epilogue flags
constexpr int F_ACC  = 1;  // accumulate into existing C
constexpr int F_BIAS = 2;  // add bias[col]
constexpr int F_SILU = 4;  // silu activation
constexpr int F_RSC  = 8;  // multiply by rowscale[row]

__device__ __forceinline__ float fast_silu(float x)
{
    return x * __builtin_amdgcn_rcpf(1.0f + __expf(-x));
}

__device__ __forceinline__ void atomic_add_f32(float* p, float v)
{
    __hip_atomic_fetch_add(p, v, __ATOMIC_RELAXED, __HIP_MEMORY_SCOPE_AGENT);
}

// Load one 16x32 WMMA operand fragment slice for this lane: elements
// K = off..off+7 and off+16..off+23 of a row. AT = float (cvt) or _Float16.
template<typename AT>
__device__ __forceinline__ v16h load_frag(const AT* __restrict__ p)
{
    v16h r;
#pragma unroll
    for (int i = 0; i < 8; ++i) {
        r[i]     = (_Float16)p[i];
        r[i + 8] = (_Float16)p[16 + i];
    }
    return r;
}

// ---------------------------------------------------------------------------
// C[M,Np] = A[M,K] @ W[Np,K]^T via v_wmma_f32_16x16x32_f16.
// W is pre-converted f16. One wave32 -> 16x32 output tile (two N-subtiles
// share the A fragment). blockDim=(32,4): block covers 16 rows x 128 cols.
// K, Np, FLAGS compile-time: fully unrolled, immediate-offset epilogue.
// Requires M%16==0, Np%128==0, K%32==0.
// ---------------------------------------------------------------------------
template<typename AT, int K, int Np, int FLAGS>
__global__ __launch_bounds__(128)
void gemm_wmma(const AT* __restrict__ A, const _Float16* __restrict__ W,
               const float* __restrict__ bias, const float* __restrict__ rscale,
               float* C, int M)
{
    const int lane = threadIdx.x;          // 0..31
    const int m16  = lane & 15;
    const int hi   = lane >> 4;
    const int tm   = blockIdx.x;
    const int tn   = (blockIdx.y * 4 + threadIdx.y) * 2;   // two 16-col tiles

    const int rowA = tm * 16 + m16;        // A row this lane streams
    const int col0 = tn * 16 + m16;        // output col / W row (subtile 0)
    const int col1 = col0 + 16;            // subtile 1

    const AT*       ap = A + (size_t)rowA * K;
    const _Float16* w0 = W + (size_t)col0 * K;
    const _Float16* w1 = W + (size_t)col1 * K;

    v8f acc0 = {}, acc1 = {};
#pragma unroll
    for (int kk = 0; kk < K; kk += 32) {
        const int ko = kk + hi * 8;        // K = ko..+7 and ko+16..+23
        v16h av  = load_frag(ap + ko);
        v16h bv0 = load_frag(w0 + ko);
        v16h bv1 = load_frag(w1 + ko);
        acc0 = __builtin_amdgcn_wmma_f32_16x16x32_f16(
                   false, av, false, bv0, (short)0, acc0, false, false);
        acc1 = __builtin_amdgcn_wmma_f32_16x16x32_f16(
                   false, av, false, bv1, (short)0, acc1, false, false);
    }

    const int rowbase = tm * 16 + hi * 8;
    float* cp = C + (size_t)rowbase * Np;
#pragma unroll
    for (int r = 0; r < 8; ++r) {
        const size_t o0 = (size_t)r * Np + col0;   // Np constexpr -> immediates
        const size_t o1 = o0 + 16;
        float v0 = acc0[r], v1 = acc1[r];
        if (FLAGS & F_ACC)  { v0 += cp[o0];     v1 += cp[o1]; }
        if (FLAGS & F_BIAS) { v0 += bias[col0]; v1 += bias[col1]; }
        if (FLAGS & F_SILU) { v0 = fast_silu(v0); v1 = fast_silu(v1); }
        if (FLAGS & F_RSC)  { float rs = rscale[rowbase + r]; v0 *= rs; v1 *= rs; }
        cp[o0] = v0;
        cp[o1] = v1;
    }
}

template<typename AT, int K, int Np, int FLAGS>
static inline void gemm_launch(const AT* A, const _Float16* W, const float* bias,
                               const float* rs, float* C, int M, hipStream_t stream)
{
    dim3 g(M / 16, Np / 128), b(32, 4);
    gemm_wmma<AT, K, Np, FLAGS><<<g, b, 0, stream>>>(A, W, bias, rs, C, M);
}

// f32 -> f16 weight conversion
__global__ void cvt_f16_kernel(const float* __restrict__ src,
                               _Float16* __restrict__ dst, int n)
{
    int i = blockIdx.x * blockDim.x + threadIdx.x;
    if (i < n) dst[i] = (_Float16)src[i];
}

// ---------------------------------------------------------------------------
// Edge featurization: cutoff C, expnorm RBF (32), unit edge vector.
// ---------------------------------------------------------------------------
__global__ void edge_feat_kernel(const int* __restrict__ ei,
                                 const float* __restrict__ ew,
                                 const float* __restrict__ ev,
                                 float* __restrict__ Cbuf,
                                 float* __restrict__ attr,
                                 float* __restrict__ vhat)
{
    int e = blockIdx.x * blockDim.x + threadIdx.x;
    if (e >= kE) return;
    float d = ew[e];
    float cut = (d < 5.0f) ? 0.5f * (__cosf(d * 0.6283185307179586f) + 1.0f) : 0.0f;
    Cbuf[e] = cut;

    const float start = 0.006737946999085467f;            // exp(-5)
    const float bi    = (2.0f / 32.0f) * (1.0f - start);
    const float beta  = 1.0f / (bi * bi);
    const float step  = (1.0f - start) / 31.0f;
    float ed = __expf(-d);
#pragma unroll 8
    for (int r = 0; r < 32; ++r) {
        float t = ed - (start + r * step);
        attr[(size_t)e * kRBF + r] = cut * __expf(-beta * t * t);
    }

    int s = ei[e], dn = ei[kE + e];
    float x = ev[e * 3], y = ev[e * 3 + 1], z = ev[e * 3 + 2];
    if (s != dn) {
        float nrm = sqrtf(x * x + y * y + z * z);
        float inv = (nrm == 0.0f) ? 1.0f : 1.0f / nrm;
        x *= inv; y *= inv; z *= inv;
    }
    vhat[e * 3] = x; vhat[e * 3 + 1] = y; vhat[e * 3 + 2] = z;
}

// Gather Zcat[e] = [Z[dst[e]], Z[src[e]]]  (E x 256)
__global__ void zcat_kernel(const int* __restrict__ z, const int* __restrict__ ei,
                            const float* __restrict__ emb, float* __restrict__ Zcat)
{
    size_t idx = (size_t)blockIdx.x * blockDim.x + threadIdx.x;
    if (idx >= (size_t)kE * 256) return;
    int e = (int)(idx >> 8);
    int c = (int)(idx & 255);
    int node = (c < kH) ? ei[kE + e] : ei[e];
    Zcat[idx] = emb[(size_t)z[node] * kH + (c & (kH - 1))];
}

// msg = Zij * (W1*skew(v) + W2*sym(v) + W3*I); atomic scatter to X[dst]
__global__ void emb_scatter_kernel(const int* __restrict__ ei,
                                   const float* __restrict__ vhat,
                                   const float* __restrict__ W1,
                                   const float* __restrict__ W2,
                                   const float* __restrict__ W3,
                                   const float* __restrict__ Zij,
                                   float* X)
{
    int e = blockIdx.x, h = threadIdx.x;
    size_t eh = (size_t)e * kH + h;
    float zv = Zij[eh];
    float a1 = zv * W1[eh], a2 = zv * W2[eh], a3 = zv * W3[eh];
    float x = vhat[e * 3], y = vhat[e * 3 + 1], z = vhat[e * 3 + 2];
    float dm = (x * x + y * y + z * z) * (1.0f / 3.0f);
    float sk[9] = {0.f, -z,  y,  z, 0.f, -x, -y,  x, 0.f};
    float sy[9] = {x * x - dm, x * y, x * z,
                   y * x, y * y - dm, y * z,
                   z * x, z * y, z * z - dm};
    float* xp = X + (size_t)ei[kE + e] * 9 * kH + h;
#pragma unroll
    for (int ij = 0; ij < 9; ++ij) {
        float idv = (ij == 0 || ij == 4 || ij == 8) ? 1.0f : 0.0f;
        atomic_add_f32(xp + (size_t)ij * kH, a1 * sk[ij] + a2 * sy[ij] + a3 * idv);
    }
}

// Decompose T -> I, A, S (f16, channel-last); optional tensor_norm output.
__global__ void decomp_kernel(const float* __restrict__ X,
                              _Float16* __restrict__ Ib, _Float16* __restrict__ Ab,
                              _Float16* __restrict__ Sb, float* tnorm)
{
    int n = blockIdx.x, h = threadIdx.x;
    size_t base = (size_t)n * 9 * kH + h;
    float t[9];
#pragma unroll
    for (int i = 0; i < 9; ++i) t[i] = X[base + (size_t)i * kH];
    float dm = (t[0] + t[4] + t[8]) * (1.0f / 3.0f);
    float s2 = 0.0f;
#pragma unroll
    for (int i = 0; i < 3; ++i)
#pragma unroll
        for (int j = 0; j < 3; ++j) {
            int ij = i * 3 + j, ji = j * 3 + i;
            float dv = (i == j) ? dm : 0.0f;
            Ib[base + (size_t)ij * kH] = (_Float16)dv;
            Ab[base + (size_t)ij * kH] = (_Float16)(0.5f * (t[ij] - t[ji]));
            Sb[base + (size_t)ij * kH] = (_Float16)(0.5f * (t[ij] + t[ji]) - dv);
            s2 += t[ij] * t[ij];
        }
    if (tnorm) tnorm[(size_t)n * kH + h] = s2;
}

// X <- X/(||X||^2+1), then decompose into I,A,S (f16)
__global__ void normdecomp_kernel(float* X, _Float16* __restrict__ Ib,
                                  _Float16* __restrict__ Ab, _Float16* __restrict__ Sb)
{
    int n = blockIdx.x, h = threadIdx.x;
    size_t base = (size_t)n * 9 * kH + h;
    float t[9];
    float s2 = 0.0f;
#pragma unroll
    for (int i = 0; i < 9; ++i) { t[i] = X[base + (size_t)i * kH]; s2 += t[i] * t[i]; }
    float inv = 1.0f / (s2 + 1.0f);
#pragma unroll
    for (int i = 0; i < 9; ++i) { t[i] *= inv; X[base + (size_t)i * kH] = t[i]; }
    float dm = (t[0] + t[4] + t[8]) * (1.0f / 3.0f);
#pragma unroll
    for (int i = 0; i < 3; ++i)
#pragma unroll
        for (int j = 0; j < 3; ++j) {
            int ij = i * 3 + j, ji = j * 3 + i;
            float dv = (i == j) ? dm : 0.0f;
            Ib[base + (size_t)ij * kH] = (_Float16)dv;
            Ab[base + (size_t)ij * kH] = (_Float16)(0.5f * (t[ij] - t[ji]));
            Sb[base + (size_t)ij * kH] = (_Float16)(0.5f * (t[ij] + t[ji]) - dv);
        }
}

// Row LayerNorm, F features (F multiple of 32, <= 384), blockDim.x == F.
__global__ void layernorm_kernel(const float* __restrict__ x,
                                 const float* __restrict__ g,
                                 const float* __restrict__ b,
                                 float* __restrict__ y, int F)
{
    __shared__ float ssum[12], ssq[12];
    __shared__ float stats[2];
    int n = blockIdx.x, t = threadIdx.x;
    float v = x[(size_t)n * F + t];
    float s = v, q = v * v;
#pragma unroll
    for (int m = 16; m >= 1; m >>= 1) {
        s += __shfl_xor(s, m, 32);
        q += __shfl_xor(q, m, 32);
    }
    if ((t & 31) == 0) { ssum[t >> 5] = s; ssq[t >> 5] = q; }
    __syncthreads();
    if (t == 0) {
        int nw = F >> 5;
        float ts = 0.0f, tq = 0.0f;
        for (int i = 0; i < nw; ++i) { ts += ssum[i]; tq += ssq[i]; }
        float mean = ts / F;
        float var  = tq / F - mean * mean;
        stats[0] = mean; stats[1] = rsqrtf(var + 1e-5f);
    }
    __syncthreads();
    y[(size_t)n * F + t] = (v - stats[0]) * stats[1] * g[t] + b[t];
}

// X = new_radial_tensor(T, f)  with f packed (n, 3h+c) in f3 (N x 3H)
__global__ void nrt_node_kernel(const float* __restrict__ T,
                                const float* __restrict__ f3, float* __restrict__ Xo)
{
    int n = blockIdx.x, h = threadIdx.x;
    size_t base = (size_t)n * 9 * kH + h;
    float t[9];
#pragma unroll
    for (int i = 0; i < 9; ++i) t[i] = T[base + (size_t)i * kH];
    size_t fb = (size_t)n * 3 * kH + 3 * h;
    float f0 = f3[fb], f1 = f3[fb + 1], f2 = f3[fb + 2];
    float dm = (t[0] + t[4] + t[8]) * (1.0f / 3.0f);
    float c0 = (f0 - f2) * dm, cp = 0.5f * (f1 + f2), cm = 0.5f * (f1 - f2);
#pragma unroll
    for (int i = 0; i < 3; ++i)
#pragma unroll
        for (int j = 0; j < 3; ++j) {
            int ij = i * 3 + j, ji = j * 3 + i;
            float dv = (i == j) ? c0 : 0.0f;
            Xo[base + (size_t)ij * kH] = dv + cp * t[ij] - cm * t[ji];
        }
}

// scatter: Msg[dst] += new_radial_tensor(Y[src], fij[e])
__global__ void msg_scatter_kernel(const int* __restrict__ ei,
                                   const float* __restrict__ Y,
                                   const float* __restrict__ fij, float* Msg)
{
    int e = blockIdx.x, h = threadIdx.x;
    int s = ei[e], dst = ei[kE + e];
    size_t yb = (size_t)s * 9 * kH + h;
    float t[9];
#pragma unroll
    for (int i = 0; i < 9; ++i) t[i] = Y[yb + (size_t)i * kH];
    size_t fb = (size_t)e * 3 * kH + 3 * h;
    float f0 = fij[fb], f1 = fij[fb + 1], f2 = fij[fb + 2];
    float dm = (t[0] + t[4] + t[8]) * (1.0f / 3.0f);
    float c0 = (f0 - f2) * dm, cp = 0.5f * (f1 + f2), cm = 0.5f * (f1 - f2);
    float* mp = Msg + (size_t)dst * 9 * kH + h;
#pragma unroll
    for (int i = 0; i < 3; ++i)
#pragma unroll
        for (int j = 0; j < 3; ++j) {
            int ij = i * 3 + j, ji = j * 3 + i;
            float dv = (i == j) ? c0 : 0.0f;
            atomic_add_f32(mp + (size_t)ij * kH, dv + cp * t[ij] - cm * t[ji]);
        }
}

// AB = M@Y + Y@M (3x3 per (n,h)); Y2 = AB/(||AB||^2+1)  (f16 out)
__global__ void abnorm_kernel(const float* __restrict__ Msg,
                              const float* __restrict__ Y, _Float16* __restrict__ Y2)
{
    int n = blockIdx.x, h = threadIdx.x;
    size_t base = (size_t)n * 9 * kH + h;
    float m[9], y[9], ab[9];
#pragma unroll
    for (int i = 0; i < 9; ++i) { m[i] = Msg[base + (size_t)i * kH]; y[i] = Y[base + (size_t)i * kH]; }
    float s2 = 0.0f;
#pragma unroll
    for (int i = 0; i < 3; ++i)
#pragma unroll
        for (int j = 0; j < 3; ++j) {
            float acc = 0.0f;
#pragma unroll
            for (int k = 0; k < 3; ++k)
                acc += m[i * 3 + k] * y[k * 3 + j] + y[i * 3 + k] * m[k * 3 + j];
            ab[i * 3 + j] = acc;
            s2 += acc * acc;
        }
    float inv = 1.0f / (s2 + 1.0f);
#pragma unroll
    for (int i = 0; i < 9; ++i) Y2[base + (size_t)i * kH] = (_Float16)(ab[i] * inv);
}

// X += dX + dX@dX
__global__ void dxupdate_kernel(const float* __restrict__ dX, float* X)
{
    int n = blockIdx.x, h = threadIdx.x;
    size_t base = (size_t)n * 9 * kH + h;
    float d[9];
#pragma unroll
    for (int i = 0; i < 9; ++i) d[i] = dX[base + (size_t)i * kH];
#pragma unroll
    for (int i = 0; i < 3; ++i)
#pragma unroll
        for (int j = 0; j < 3; ++j) {
            float acc = d[i * 3 + j];
#pragma unroll
            for (int k = 0; k < 3; ++k) acc += d[i * 3 + k] * d[k * 3 + j];
            X[base + (size_t)(i * 3 + j) * kH] += acc;
        }
}

// feat = [||I||^2, ||A||^2, ||S||^2]  -> (N, 3H)
__global__ void head_kernel(const float* __restrict__ X, float* __restrict__ feat)
{
    int n = blockIdx.x, h = threadIdx.x;
    size_t base = (size_t)n * 9 * kH + h;
    float t[9];
#pragma unroll
    for (int i = 0; i < 9; ++i) t[i] = X[base + (size_t)i * kH];
    float dm = (t[0] + t[4] + t[8]) * (1.0f / 3.0f);
    float nI = 3.0f * dm * dm, nA = 0.0f, nS = 0.0f;
#pragma unroll
    for (int i = 0; i < 3; ++i)
#pragma unroll
        for (int j = 0; j < 3; ++j) {
            int ij = i * 3 + j, ji = j * 3 + i;
            float av = 0.5f * (t[ij] - t[ji]);
            float sv = 0.5f * (t[ij] + t[ji]) - ((i == j) ? dm : 0.0f);
            nA += av * av;
            nS += sv * sv;
        }
    size_t fb = (size_t)n * 3 * kH + h;
    feat[fb] = nI; feat[fb + kH] = nA; feat[fb + 2 * kH] = nS;
}

// ---------------------------------------------------------------------------
extern "C" void kernel_launch(void* const* d_in, const int* in_sizes, int n_in,
                              void* d_out, int out_size, void* d_ws, size_t ws_size,
                              hipStream_t stream)
{
    (void)in_sizes; (void)n_in; (void)out_size; (void)ws_size;

    const int*   z        = (const int*)  d_in[0];
    const int*   ei       = (const int*)  d_in[1];
    const float* ew       = (const float*)d_in[2];
    const float* ev       = (const float*)d_in[3];
    const float* emb_w    = (const float*)d_in[4];
    const float* dp1_w    = (const float*)d_in[5];
    const float* dp1_b    = (const float*)d_in[6];
    const float* dp2_w    = (const float*)d_in[7];
    const float* dp2_b    = (const float*)d_in[8];
    const float* dp3_w    = (const float*)d_in[9];
    const float* dp3_b    = (const float*)d_in[10];
    const float* emb2_w   = (const float*)d_in[11];
    const float* emb2_b   = (const float*)d_in[12];
    const float* te_lt_w  = (const float*)d_in[13];
    const float* te_ls0_w = (const float*)d_in[14];
    const float* te_ls0_b = (const float*)d_in[15];
    const float* te_ls1_w = (const float*)d_in[16];
    const float* te_ls1_b = (const float*)d_in[17];
    const float* in_g     = (const float*)d_in[18];
    const float* in_b     = (const float*)d_in[19];
    const float* ml0_w    = (const float*)d_in[20];
    const float* ml0_b    = (const float*)d_in[21];
    const float* ml1_w    = (const float*)d_in[22];
    const float* ml1_b    = (const float*)d_in[23];
    const float* ml2_w    = (const float*)d_in[24];
    const float* ml2_b    = (const float*)d_in[25];
    const float* mlI_w    = (const float*)d_in[26];
    const float* mlA_w    = (const float*)d_in[27];
    const float* mlS_w    = (const float*)d_in[28];
    const float* mlY_w    = (const float*)d_in[29];
    const float* ml_lt_w  = (const float*)d_in[30];
    const float* out_g    = (const float*)d_in[31];
    const float* out_bn   = (const float*)d_in[32];
    const float* out_w    = (const float*)d_in[33];
    const float* out_b    = (const float*)d_in[34];

    char* wsb = (char*)d_ws;
    size_t off = 0;
    auto allocf = [&](size_t n) {
        float* p = (float*)(wsb + off);
        off = (off + n * 4 + 255) & ~(size_t)255;
        return p;
    };
    auto alloch = [&](size_t n) {
        _Float16* p = (_Float16*)(wsb + off);
        off = (off + n * 2 + 255) & ~(size_t)255;
        return p;
    };

    float* Cbuf  = allocf(kE);
    float* attr  = allocf((size_t)kE * kRBF);
    float* vhat  = allocf((size_t)kE * 3);
    float* Wbuf  = allocf((size_t)kE * 3 * kH);   // W1|W2|W3; reused as fij (E,3H)
    float* Zcat  = allocf((size_t)kE * 2 * kH);   // reused as e2 (E,2H)
    float* Zij   = allocf((size_t)kE * kH);       // reused as e1 (E,H)
    float* X     = allocf(kN9H);
    float* Ybuf  = allocf(kN9H);
    float* Mbuf  = allocf(kN9H);                  // msg; reused as dX
    float* tnorm = allocf((size_t)kN * kH);
    float* nb1   = allocf((size_t)kN * kH);
    float* nb2   = allocf((size_t)kN * 2 * kH);
    float* nb3   = allocf((size_t)kN * 3 * kH);
    float* lnbuf = allocf((size_t)kN * 3 * kH);

    _Float16* Ib = alloch(kN9H);
    _Float16* Ab = alloch(kN9H);
    _Float16* Sb = alloch(kN9H);
    _Float16* Y2 = alloch(kN9H);

    // f16 weight copies
    _Float16* dp1h  = alloch(kH * kRBF);
    _Float16* dp2h  = alloch(kH * kRBF);
    _Float16* dp3h  = alloch(kH * kRBF);
    _Float16* emb2h = alloch(kH * 2 * kH);
    _Float16* telth = alloch(3 * kH * kH);
    _Float16* tel0h = alloch(2 * kH * kH);
    _Float16* tel1h = alloch(3 * kH * 2 * kH);
    _Float16* ml0h  = alloch(2 * kH * kRBF);
    _Float16* ml1h  = alloch(2 * 2 * kH * kH);
    _Float16* ml2h  = alloch(2 * 3 * kH * 2 * kH);
    _Float16* mlIh  = alloch(2 * kH * kH);
    _Float16* mlAh  = alloch(2 * kH * kH);
    _Float16* mlSh  = alloch(2 * kH * kH);
    _Float16* mlYh  = alloch(2 * kH * kH);
    _Float16* mllth = alloch(2 * 3 * kH * kH);
    _Float16* outwh = alloch(kH * 3 * kH);

    float* W1 = Wbuf;
    float* W2 = Wbuf + (size_t)kE * kH;
    float* W3 = Wbuf + (size_t)2 * kE * kH;
    float* e1 = Zij;
    float* e2 = Zcat;
    float* fij = Wbuf;

    auto cvt = [&](const float* src, _Float16* dst, int n) {
        cvt_f16_kernel<<<(n + 255) / 256, 256, 0, stream>>>(src, dst, n);
    };
    cvt(dp1_w, dp1h, kH * kRBF);
    cvt(dp2_w, dp2h, kH * kRBF);
    cvt(dp3_w, dp3h, kH * kRBF);
    cvt(emb2_w, emb2h, kH * 2 * kH);
    cvt(te_lt_w, telth, 3 * kH * kH);
    cvt(te_ls0_w, tel0h, 2 * kH * kH);
    cvt(te_ls1_w, tel1h, 3 * kH * 2 * kH);
    cvt(ml0_w, ml0h, 2 * kH * kRBF);
    cvt(ml1_w, ml1h, 2 * 2 * kH * kH);
    cvt(ml2_w, ml2h, 2 * 3 * kH * 2 * kH);
    cvt(mlI_w, mlIh, 2 * kH * kH);
    cvt(mlA_w, mlAh, 2 * kH * kH);
    cvt(mlS_w, mlSh, 2 * kH * kH);
    cvt(mlY_w, mlYh, 2 * kH * kH);
    cvt(ml_lt_w, mllth, 2 * 3 * kH * kH);
    cvt(out_w, outwh, kH * 3 * kH);

    // ---- edge featurization ----
    edge_feat_kernel<<<(kE + 255) / 256, 256, 0, stream>>>(ei, ew, ev, Cbuf, attr, vhat);

    // ---- TensorNeighborEmbedding ----
    gemm_launch<float, kRBF, kH, F_BIAS | F_RSC>(attr, dp1h, dp1_b, Cbuf, W1, kE, stream);
    gemm_launch<float, kRBF, kH, F_BIAS | F_RSC>(attr, dp2h, dp2_b, Cbuf, W2, kE, stream);
    gemm_launch<float, kRBF, kH, F_BIAS | F_RSC>(attr, dp3h, dp3_b, Cbuf, W3, kE, stream);

    zcat_kernel<<<(unsigned)(((size_t)kE * 256 + 255) / 256), 256, 0, stream>>>(z, ei, emb_w, Zcat);
    gemm_launch<float, 2 * kH, kH, F_BIAS>(Zcat, emb2h, emb2_b, nullptr, Zij, kE, stream);

    hipMemsetAsync(X, 0, (size_t)kN9H * sizeof(float), stream);
    emb_scatter_kernel<<<kE, kH, 0, stream>>>(ei, vhat, W1, W2, W3, Zij, X);

    decomp_kernel<<<kN, kH, 0, stream>>>(X, Ib, Ab, Sb, tnorm);
    layernorm_kernel<<<kN, kH, 0, stream>>>(tnorm, in_g, in_b, nb1, kH);

    gemm_launch<_Float16, kH, kH, 0>    (Ib, telth + 0 * kH * kH, nullptr, nullptr, Ybuf, kN * 9, stream);
    gemm_launch<_Float16, kH, kH, F_ACC>(Ab, telth + 1 * kH * kH, nullptr, nullptr, Ybuf, kN * 9, stream);
    gemm_launch<_Float16, kH, kH, F_ACC>(Sb, telth + 2 * kH * kH, nullptr, nullptr, Ybuf, kN * 9, stream);

    gemm_launch<float, kH, 2 * kH, F_BIAS | F_SILU>(nb1, tel0h, te_ls0_b, nullptr, nb2, kN, stream);
    gemm_launch<float, 2 * kH, 3 * kH, F_BIAS | F_SILU>(nb2, tel1h, te_ls1_b, nullptr, nb3, kN, stream);

    nrt_node_kernel<<<kN, kH, 0, stream>>>(Ybuf, nb3, X);

    // ---- TensorMessage layers ----
    for (int l = 0; l < 2; ++l) {
        const _Float16* m0w = ml0h + (size_t)l * kH * kRBF;
        const float*    m0b = ml0_b + (size_t)l * kH;
        const _Float16* m1w = ml1h + (size_t)l * 2 * kH * kH;
        const float*    m1b = ml1_b + (size_t)l * 2 * kH;
        const _Float16* m2w = ml2h + (size_t)l * 3 * kH * 2 * kH;
        const float*    m2b = ml2_b + (size_t)l * 3 * kH;
        const _Float16* iw  = mlIh + (size_t)l * kH * kH;
        const _Float16* aw  = mlAh + (size_t)l * kH * kH;
        const _Float16* sw  = mlSh + (size_t)l * kH * kH;
        const _Float16* yw  = mlYh + (size_t)l * kH * kH;
        const _Float16* ltw = mllth + (size_t)l * 3 * kH * kH;

        gemm_launch<float, kRBF, kH, F_BIAS | F_SILU>(attr, m0w, m0b, nullptr, e1, kE, stream);
        gemm_launch<float, kH, 2 * kH, F_BIAS | F_SILU>(e1, m1w, m1b, nullptr, e2, kE, stream);
        gemm_launch<float, 2 * kH, 3 * kH, F_BIAS | F_SILU | F_RSC>(e2, m2w, m2b, Cbuf, fij, kE, stream);

        normdecomp_kernel<<<kN, kH, 0, stream>>>(X, Ib, Ab, Sb);

        gemm_launch<_Float16, kH, kH, 0>    (Ib, iw, nullptr, nullptr, Ybuf, kN * 9, stream);
        gemm_launch<_Float16, kH, kH, F_ACC>(Ab, aw, nullptr, nullptr, Ybuf, kN * 9, stream);
        gemm_launch<_Float16, kH, kH, F_ACC>(Sb, sw, nullptr, nullptr, Ybuf, kN * 9, stream);

        hipMemsetAsync(Mbuf, 0, (size_t)kN9H * sizeof(float), stream);
        msg_scatter_kernel<<<kE, kH, 0, stream>>>(ei, Ybuf, fij, Mbuf);

        abnorm_kernel<<<kN, kH, 0, stream>>>(Mbuf, Ybuf, Y2);

        gemm_launch<_Float16, kH, kH, 0>(Y2, yw, nullptr, nullptr, Ybuf, kN * 9, stream);
        decomp_kernel<<<kN, kH, 0, stream>>>(Ybuf, Ib, Ab, Sb, nullptr);

        gemm_launch<_Float16, kH, kH, 0>    (Ib, ltw + 0 * kH * kH, nullptr, nullptr, Mbuf, kN * 9, stream);
        gemm_launch<_Float16, kH, kH, F_ACC>(Ab, ltw + 1 * kH * kH, nullptr, nullptr, Mbuf, kN * 9, stream);
        gemm_launch<_Float16, kH, kH, F_ACC>(Sb, ltw + 2 * kH * kH, nullptr, nullptr, Mbuf, kN * 9, stream);

        dxupdate_kernel<<<kN, kH, 0, stream>>>(Mbuf, X);
    }

    // ---- output head ----
    head_kernel<<<kN, kH, 0, stream>>>(X, nb3);
    layernorm_kernel<<<kN, 3 * kH, 0, stream>>>(nb3, out_g, out_bn, lnbuf, 3 * kH);
    gemm_launch<float, 3 * kH, kH, F_BIAS | F_SILU>(lnbuf, outwh, out_b, nullptr, (float*)d_out, kN, stream);
}
